// AutoCorrelationLayer_31275951849558
// MI455X (gfx1250) — matrix operand order repaired and checked by reading, not compiled, not run on hardware
//
#include <hip/hip_runtime.h>
#include <hip/hip_bf16.h>
#include <math.h>

#define B_    8
#define L_    2048
#define D_    1024
#define H_    16
#define DK_   64
#define TOPK_ 7
#define M_    (B_*L_)   // 16384 rows

typedef _Float16 half_t;
typedef __attribute__((ext_vector_type(16))) _Float16 v16h;
typedef __attribute__((ext_vector_type(8)))  _Float16 v8h;
typedef __attribute__((ext_vector_type(4)))  _Float16 v4h;
typedef __attribute__((ext_vector_type(8)))  float    v8f;
typedef __attribute__((ext_vector_type(4)))  float    v4f;

// CDNA5 async global->LDS path (ASYNCcnt), via inline asm (the ROCm builtin
// wants pointers in the unspellable __device__/__shared__ LangAS). Guarded so
// the host pass never sees gfx1250 asm; sync fallback kept for safety.
#if defined(__HIP_DEVICE_COMPILE__) && defined(__gfx1250__)
#define AC_ASYNC 1
#else
#define AC_ASYNC 0
#endif

#if AC_ASYNC
static __device__ __forceinline__ void async_ld_b128(const void* g, void* l) {
  // VDST = per-lane LDS byte address (flat->LDS uses addr[31:0]);
  // VADDR = 64-bit global address; GV mode (saddr = off).
  unsigned lds_off = (unsigned)(size_t)l;
  asm volatile("global_load_async_to_lds_b128 %0, %1, off"
               :: "v"(lds_off), "v"(g)
               : "memory");
}
static __device__ __forceinline__ void wait_async_0() {
  asm volatile("s_wait_asynccnt 0x0" ::: "memory");
}
static __device__ __forceinline__ void wait_async_1() {
  asm volatile("s_wait_asynccnt 0x1" ::: "memory");
}
#endif

// ---- WMMA helper: D = A(16x32 f16) * B(32x16 f16) + C(f32) ----
static __device__ __forceinline__ v8f wmma_f16(v16h a, v16h b, v8f c) {
  return __builtin_amdgcn_wmma_f32_16x16x32_f16(
      /*neg_a=*/false, a, /*neg_b=*/false, b,
      /*c_mod=*/(short)0, c, /*reuse_a=*/false, /*reuse_b=*/false);
}

// A-fragment (16x32, M x K), row-major f16 source.
// Lane = l16 + 16*hi holds row M=l16, K in {k0+hi*8..+7} and {k0+16+hi*8..+7}.
static __device__ __forceinline__ v16h load_a_f16(const half_t* base, int row0,
                                                  int ld, int k0, int lane) {
  int l16 = lane & 15, hi = lane >> 4;
  const half_t* p = base + (size_t)(row0 + l16) * ld + k0 + hi * 8;
  v8h lo = *(const v8h*)p;
  v8h hh = *(const v8h*)(p + 16);
  v16h r;
#pragma unroll
  for (int i = 0; i < 8; ++i) { r[i] = lo[i]; r[i + 8] = hh[i]; }
  return r;
}

// Same but converting from f32 source on the fly.
static __device__ __forceinline__ v16h load_a_f32(const float* base, int row0,
                                                  int ld, int k0, int lane) {
  int l16 = lane & 15, hi = lane >> 4;
  const float* p = base + (size_t)(row0 + l16) * ld + k0 + hi * 8;
  v16h r;
#pragma unroll
  for (int i = 0; i < 8; ++i) {
    r[i]     = (half_t)p[i];
    r[i + 8] = (half_t)p[16 + i];
  }
  return r;
}

// B-fragment (32x16, K x N) with B[k,n] = src[row0+n][k] (NT GEMM).
// Lane n(+16*hi) holds K in {k0+hi*8..+7} and {k0+16+hi*8..+7}.
static __device__ __forceinline__ v16h load_b_f16(const half_t* base, int row0,
                                                  int ld, int k0, int lane) {
  int l16 = lane & 15, hi = lane >> 4;
  const half_t* p = base + (size_t)(row0 + l16) * ld + k0 + hi * 8;
  v8h lo = *(const v8h*)p;
  v8h hh = *(const v8h*)(p + 16);
  v16h r;
#pragma unroll
  for (int i = 0; i < 8; ++i) { r[i] = lo[i]; r[i + 8] = hh[i]; }
  return r;
}

// ---------------- weight f32 -> f16 convert ----------------
__global__ __launch_bounds__(256) void ac_cvt_kernel(const float* __restrict__ src,
                                                     half_t* __restrict__ dst) {
  int i = (blockIdx.x * 256 + threadIdx.x) * 4;   // 1024 blocks -> 1M elems
  v4f x = *(const v4f*)(src + i);
  v4h o;
#pragma unroll
  for (int j = 0; j < 4; ++j) o[j] = (half_t)x[j];
  *(v4h*)(dst + i) = o;
}

// ---------------- fused Q/K/V projection (NT GEMM, WMMA) ----------------
// Block = 128(M) x 64(N) tile, 8 waves of 16x64. Weight slab (64x32 f16)
// double-buffered in LDS via async global->LDS DMA, shared by all waves.
// y[m,c] = sum_k X[m,k]*W[c,k] + bias[c]; store per-head (b,h,l,dk).
__global__ __launch_bounds__(256) void ac_proj_kernel(
    const float* __restrict__ xq, const float* __restrict__ xk, const float* __restrict__ xv,
    const half_t* __restrict__ wq16, const half_t* __restrict__ wk16, const half_t* __restrict__ wv16,
    const float* __restrict__ bq, const float* __restrict__ bk, const float* __restrict__ bv,
    half_t* __restrict__ qh, half_t* __restrict__ kh, float* __restrict__ vh) {
  __shared__ __align__(128) half_t bs[2][64 * 32];   // 2 x 4KB weight tiles

  const int tid  = threadIdx.x;
  const int lane = tid & 31;
  const int wave = tid >> 5;
  const int nblk = blockIdx.x & 15;       // 16 n-blocks of 64
  const int mblk = blockIdx.x >> 4;       // 128 m-blocks of 128
  const int n0 = nblk * 64;
  const int m0 = mblk * 128 + wave * 16;
  const int sel = blockIdx.y;             // 0:q 1:k 2:v (uniform)
  const float*  X    = sel == 0 ? xq   : (sel == 1 ? xk   : xv);
  const half_t* W    = sel == 0 ? wq16 : (sel == 1 ? wk16 : wv16);
  const float*  bias = sel == 0 ? bq   : (sel == 1 ? bk   : bv);

  // Per-thread 16B chunk of the 64x32 weight tile: row = tid>>2, col8 = (tid&3)*8
  const int brow = tid >> 2, bcol = (tid & 3) * 8;
  const half_t* wsrc = W + (size_t)(n0 + brow) * D_ + bcol;
  half_t* lb0 = &bs[0][brow * 32 + bcol];
  half_t* lb1 = &bs[1][brow * 32 + bcol];

  v8f acc[4] = {};
#if AC_ASYNC
  async_ld_b128(wsrc, lb0);
  int cur = 0;
  for (int k0 = 0; k0 < D_; k0 += 32) {
    if (k0 + 32 < D_) {
      async_ld_b128(wsrc + k0 + 32, cur ? lb0 : lb1);
      wait_async_1();                  // previous tile landed in LDS
    } else {
      wait_async_0();
    }
    __syncthreads();                   // all waves' DMA complete
    __builtin_prefetch(X + (size_t)(m0 + (lane & 15)) * D_ + k0 + 64, 0, 0);
    v16h a = load_a_f32(X, m0, D_, k0, lane);
#pragma unroll
    for (int j = 0; j < 4; ++j) {
      v16h bf = load_b_f16(&bs[cur][0], 16 * j, 32, 0, lane);
      acc[j] = wmma_f16(a, bf, acc[j]);
    }
    __syncthreads();                   // reads done; buffer reusable
    cur ^= 1;
  }
#else
  for (int k0 = 0; k0 < D_; k0 += 32) {
    v8h t = *(const v8h*)(wsrc + k0);
    __syncthreads();
    *(v8h*)lb0 = t;
    __syncthreads();
    v16h a = load_a_f32(X, m0, D_, k0, lane);
#pragma unroll
    for (int j = 0; j < 4; ++j) {
      v16h bf = load_b_f16(&bs[0][0], 16 * j, 32, 0, lane);
      acc[j] = wmma_f16(a, bf, acc[j]);
    }
  }
  (void)lb1;
#endif

  // C/D layout: VGPR r -> M = r + 8*hi, lane -> N = l16.
  const int l16 = lane & 15, hi = lane >> 4;
  const int b = m0 >> 11;
  const int lbase = (m0 & (L_ - 1)) + 8 * hi;
#pragma unroll
  for (int j = 0; j < 4; ++j) {
    int c = n0 + 16 * j + l16;
    int h = c >> 6, dk = c & 63;
    float bb = bias[c];
    size_t base = ((size_t)(b * H_ + h) * L_) * DK_ + dk;
#pragma unroll
    for (int r = 0; r < 8; ++r) {
      float val = acc[j][r] + bb;
      size_t addr = base + (size_t)(lbase + r) * DK_;
      if (sel == 0)      qh[addr] = (half_t)val;
      else if (sel == 1) kh[addr] = (half_t)val;
      else               vh[addr] = val;
    }
  }
}

// ---------------- correlation: mean_value[tau] = (1/DK) sum_t <q_t, k_{t-tau}> ----
// One workgroup per (b,h). K head (2048x64 f16 = 256KB) DMA'd into LDS,
// Gram tiles via WMMA, circular-diagonal reduction via ds_add_f32.
__global__ __launch_bounds__(256) void ac_corr_kernel(const half_t* __restrict__ qh,
                                                      const half_t* __restrict__ kh,
                                                      float* __restrict__ meanout) {
  extern __shared__ char smem[];
  half_t* ks   = (half_t*)smem;                          // [2048][64] f16
  float*  msum = (float*)(smem + (size_t)L_ * DK_ * 2);  // [2048] f32

  const int bh = blockIdx.x;
  const half_t* qhead = qh + (size_t)bh * L_ * DK_;
  const half_t* khead = kh + (size_t)bh * L_ * DK_;

#if AC_ASYNC
  for (int i = threadIdx.x * 8; i < L_ * DK_; i += 2048)
    async_ld_b128(khead + i, ks + i);
#else
  for (int i = threadIdx.x * 8; i < L_ * DK_; i += 2048)
    *(v8h*)(ks + i) = *(const v8h*)(khead + i);
#endif
  for (int i = threadIdx.x; i < L_; i += 256) msum[i] = 0.0f;
#if AC_ASYNC
  wait_async_0();
#endif
  __syncthreads();

  const int wave = threadIdx.x >> 5;
  const int lane = threadIdx.x & 31;
  const int l16 = lane & 15, hi = lane >> 4;

  for (int tt = wave; tt < L_ / 16; tt += 8) {
    const int t0 = tt * 16;
    v16h a0 = load_a_f16(qhead, t0, DK_, 0,  lane);
    v16h a1 = load_a_f16(qhead, t0, DK_, 32, lane);
    for (int ss = 0; ss < L_ / 16; ++ss) {
      const int s0 = ss * 16;
      v16h b0 = load_b_f16(ks, s0, DK_, 0,  lane);
      v16h b1 = load_b_f16(ks, s0, DK_, 32, lane);
      v8f c = {};
      c = wmma_f16(a0, b0, c);
      c = wmma_f16(a1, b1, c);
      // G[t,s] -> tau = (t - s) mod L; t = t0+r+8*hi, s = s0+l16.
      const int tb = t0 - s0 + 8 * hi - l16;
#pragma unroll
      for (int r = 0; r < 8; ++r) {
        int tau = (tb + r) & (L_ - 1);
        atomicAdd(&msum[tau], c[r]);   // ds_add_f32
      }
    }
  }
  __syncthreads();
  const float inv = 1.0f / (float)DK_;
  for (int i = threadIdx.x; i < L_; i += 256)
    meanout[(size_t)bh * L_ + i] = msum[i] * inv;
}

// ---------------- top-7 + softmax per (b,h) ----------------
__global__ __launch_bounds__(256) void ac_topk_kernel(const float* __restrict__ meanv,
                                                      float* __restrict__ cw_out,
                                                      float* __restrict__ wtop,
                                                      int* __restrict__ dtop) {
  __shared__ float vals[L_];
  __shared__ float rmax[256];
  __shared__ int   ridx[256];
  __shared__ float topv[TOPK_];
  __shared__ int   topi[TOPK_];
  const int bh = blockIdx.x, tid = threadIdx.x;
  for (int i = tid; i < L_; i += 256) vals[i] = meanv[(size_t)bh * L_ + i];
  __syncthreads();

  for (int it = 0; it < TOPK_; ++it) {
    float bv = -3.402823466e38f; int bi = 0;
    for (int i = tid; i < L_; i += 256)
      if (vals[i] > bv) { bv = vals[i]; bi = i; }
    rmax[tid] = bv; ridx[tid] = bi;
    __syncthreads();
    for (int off = 128; off > 0; off >>= 1) {
      if (tid < off && rmax[tid + off] > rmax[tid]) {
        rmax[tid] = rmax[tid + off]; ridx[tid] = ridx[tid + off];
      }
      __syncthreads();
    }
    if (tid == 0) { topv[it] = rmax[0]; topi[it] = ridx[0]; vals[ridx[0]] = -3.402823466e38f; }
    __syncthreads();
  }

  if (tid == 0) {
    float mx = topv[0];
    float e[TOPK_], s = 0.0f;
    for (int i = 0; i < TOPK_; ++i) { e[i] = __expf(topv[i] - mx); s += e[i]; }
    float invs = 1.0f / s;
    for (int i = 0; i < TOPK_; ++i) {
      float w = e[i] * invs;
      cw_out[bh * TOPK_ + i] = w;       // corr_weights output (d_out tail)
      wtop[bh * TOPK_ + i]   = w;
      dtop[bh * TOPK_ + i]   = topi[i];
    }
  }
}

// ---------------- delay-gather aggregation -> agg16 (B,L,D) f16 ----------------
__global__ __launch_bounds__(256) void ac_agg_kernel(const float* __restrict__ vh,
                                                     const float* __restrict__ wtop,
                                                     const int* __restrict__ dtop,
                                                     half_t* __restrict__ agg16) {
  const int bl = blockIdx.x;            // 0..16383 = b*L + l
  const int b = bl >> 11, l = bl & (L_ - 1);
  const int h = threadIdx.x >> 4;
  const int g = threadIdx.x & 15;

  __shared__ float swt[H_][8];
  __shared__ int   sdl[H_][8];
  if (threadIdx.x < H_ * TOPK_) {
    int h2 = threadIdx.x / TOPK_, i = threadIdx.x % TOPK_;
    swt[h2][i] = wtop[(b * H_ + h2) * TOPK_ + i];
    sdl[h2][i] = dtop[(b * H_ + h2) * TOPK_ + i];
  }
  __syncthreads();

  const float* vhead = vh + (size_t)(b * H_ + h) * L_ * DK_;
  v4f acc = {};
#pragma unroll
  for (int i = 0; i < TOPK_; ++i) {
    int li = (l + sdl[h][i]) & (L_ - 1);
    v4f x = *(const v4f*)(vhead + (size_t)li * DK_ + g * 4);
    float w = swt[h][i];
    v4f wv = {w, w, w, w};
    acc += wv * x;
  }
  half_t* o = agg16 + (size_t)bl * D_ + h * DK_ + g * 4;
  v4h oh;
#pragma unroll
  for (int j = 0; j < 4; ++j) oh[j] = (half_t)acc[j];
  *(v4h*)o = oh;
}

// ---------------- output projection + bias + residual (WMMA NT GEMM) ----------------
__global__ __launch_bounds__(256) void ac_out_kernel(const half_t* __restrict__ agg16,
                                                     const half_t* __restrict__ wo16,
                                                     const float* __restrict__ bo,
                                                     const float* __restrict__ residual,
                                                     float* __restrict__ out) {
  __shared__ __align__(128) half_t bs[2][64 * 32];

  const int tid  = threadIdx.x;
  const int lane = tid & 31;
  const int wave = tid >> 5;
  const int nblk = blockIdx.x & 15;
  const int mblk = blockIdx.x >> 4;
  const int n0 = nblk * 64;
  const int m0 = mblk * 128 + wave * 16;

  const int brow = tid >> 2, bcol = (tid & 3) * 8;
  const half_t* wsrc = wo16 + (size_t)(n0 + brow) * D_ + bcol;
  half_t* lb0 = &bs[0][brow * 32 + bcol];
  half_t* lb1 = &bs[1][brow * 32 + bcol];

  v8f acc[4] = {};
#if AC_ASYNC
  async_ld_b128(wsrc, lb0);
  int cur = 0;
  for (int k0 = 0; k0 < D_; k0 += 32) {
    if (k0 + 32 < D_) {
      async_ld_b128(wsrc + k0 + 32, cur ? lb0 : lb1);
      wait_async_1();
    } else {
      wait_async_0();
    }
    __syncthreads();
    __builtin_prefetch(agg16 + (size_t)(m0 + (lane & 15)) * D_ + k0 + 64, 0, 0);
    v16h a = load_a_f16(agg16, m0, D_, k0, lane);
#pragma unroll
    for (int j = 0; j < 4; ++j) {
      v16h bf = load_b_f16(&bs[cur][0], 16 * j, 32, 0, lane);
      acc[j] = wmma_f16(a, bf, acc[j]);
    }
    __syncthreads();
    cur ^= 1;
  }
#else
  for (int k0 = 0; k0 < D_; k0 += 32) {
    v8h t = *(const v8h*)(wsrc + k0);
    __syncthreads();
    *(v8h*)lb0 = t;
    __syncthreads();
    v16h a = load_a_f16(agg16, m0, D_, k0, lane);
#pragma unroll
    for (int j = 0; j < 4; ++j) {
      v16h bf = load_b_f16(&bs[0][0], 16 * j, 32, 0, lane);
      acc[j] = wmma_f16(a, bf, acc[j]);
    }
  }
  (void)lb1;
#endif

  const int l16 = lane & 15, hi = lane >> 4;
#pragma unroll
  for (int j = 0; j < 4; ++j) {
    int c = n0 + 16 * j + l16;
    float bb = bo[c];
#pragma unroll
    for (int r = 0; r < 8; ++r) {
      int m = m0 + r + 8 * hi;
      size_t addr = (size_t)m * D_ + c;
      out[addr] = acc[j][r] + bb + residual[addr];
    }
  }
}

extern "C" void kernel_launch(void* const* d_in, const int* in_sizes, int n_in,
                              void* d_out, int out_size, void* d_ws, size_t ws_size,
                              hipStream_t stream) {
  (void)in_sizes; (void)n_in; (void)out_size; (void)ws_size;
  const float* query = (const float*)d_in[0];
  const float* key   = (const float*)d_in[1];
  const float* value = (const float*)d_in[2];
  const float* Wq = (const float*)d_in[3];
  const float* bq = (const float*)d_in[4];
  const float* Wk = (const float*)d_in[5];
  const float* bk = (const float*)d_in[6];
  const float* Wv = (const float*)d_in[7];
  const float* bv = (const float*)d_in[8];
  const float* Wo = (const float*)d_in[9];
  const float* bo = (const float*)d_in[10];

  // Scratch layout (bytes): weights f16 0..8MB | qh 8..40 | kh 40..72 |
  // vh f32 72..136 | agg16 136..168 | mean 168..169 | top-k tables.
  char* ws = (char*)d_ws;
  half_t* wq16  = (half_t*)(ws);
  half_t* wk16  = (half_t*)(ws + (size_t)(2u  << 20));
  half_t* wv16  = (half_t*)(ws + (size_t)(4u  << 20));
  half_t* wo16  = (half_t*)(ws + (size_t)(6u  << 20));
  half_t* qh    = (half_t*)(ws + (size_t)(8u  << 20));
  half_t* kh    = (half_t*)(ws + (size_t)(40u << 20));
  float*  vh    = (float* )(ws + (size_t)(72u << 20));
  half_t* agg16 = (half_t*)(ws + (size_t)(136u << 20));
  float*  meanv = (float* )(ws + (size_t)(168u << 20));
  float*  wtop  = (float* )(ws + (size_t)(169u << 20));
  int*    dtop  = (int*  )(ws + (size_t)(169u << 20) + 8192);

  float* outp = (float*)d_out;
  float* cw   = outp + (size_t)M_ * D_;    // corr_weights tail (B*H*TOPK)

  ac_cvt_kernel<<<1024, 256, 0, stream>>>(Wq, wq16);
  ac_cvt_kernel<<<1024, 256, 0, stream>>>(Wk, wk16);
  ac_cvt_kernel<<<1024, 256, 0, stream>>>(Wv, wv16);
  ac_cvt_kernel<<<1024, 256, 0, stream>>>(Wo, wo16);

  ac_proj_kernel<<<dim3(2048, 3), 256, 0, stream>>>(
      query, key, value, wq16, wk16, wv16, bq, bk, bv, qh, kh, vh);

  size_t corr_smem = (size_t)L_ * DK_ * 2 + (size_t)L_ * 4;  // 264KB < 320KB WGP LDS
  ac_corr_kernel<<<B_ * H_, 256, corr_smem, stream>>>(qh, kh, meanv);

  ac_topk_kernel<<<B_ * H_, 256, 0, stream>>>(meanv, cw, wtop, dtop);
  ac_agg_kernel<<<M_, 256, 0, stream>>>(vh, wtop, dtop, agg16);
  ac_out_kernel<<<2048, 256, 0, stream>>>(agg16, wo16, bo, query, outp);
}
